// KernelAttention_515396076424
// MI455X (gfx1250) — matrix operand order repaired and checked by reading, not compiled, not run on hardware
//
#include <hip/hip_runtime.h>
#include <hip/hip_bf16.h>

// ---------------------------------------------------------------------------
// Problem constants (from reference): B=8, C=128, H=128, WIN=4, K=3, WS=64
// ---------------------------------------------------------------------------
#define BN    8
#define CN    128
#define HN    128
#define LN    (HN * HN)       // 16384
#define WINN  4
#define WSN   64
#define KHW   9               // 3x3
#define KC    (CN * KHW)      // 1152 : contraction length of merged 3x3 conv

typedef __bf16 bf16_t;
typedef bf16_t v16bf __attribute__((ext_vector_type(16)));
typedef float  v8f   __attribute__((ext_vector_type(8)));

union FragA { v16bf v; uint4 q[2]; };
union FragB { v16bf v; uint4 q[2]; };

// ---------------------------------------------------------------------------
// Kernel 1: per-window shifted sums (mean(conv3x3) = conv1_w . shifted sums).
// Deterministic: one thread owns each (b,w,j) fully (no atomics).
// stats layout: 9 planes of [B][WIN][C]: T,rF,rL,cF,cL,c00,c0L,cL0,cLL
// ---------------------------------------------------------------------------
__global__ void stats_kernel(const float* __restrict__ x, float* __restrict__ stats) {
    const int w = blockIdx.x, b = blockIdx.y, j = threadIdx.x;
    const int wy = (w >> 1) * WSN, wx = (w & 1) * WSN;
    const float* xb = x + (size_t)b * LN * CN;
    float T = 0.f, rF = 0.f, rL = 0.f, cF = 0.f, cL = 0.f;
    float c00 = 0.f, c0L = 0.f, cL0 = 0.f, cLL = 0.f;
    for (int y = 0; y < WSN; ++y) {
        const float* row = xb + ((size_t)(wy + y) * HN + wx) * CN + j;
        float rs = 0.f, v0 = 0.f, v63 = 0.f;
        for (int xi = 0; xi < WSN; ++xi) {
            float v = row[(size_t)xi * CN];
            rs += v;
            if (xi == 0)       v0  = v;
            if (xi == WSN - 1) v63 = v;
        }
        T += rs; cF += v0; cL += v63;
        if (y == 0)       { rF = rs; c00 = v0; c0L = v63; }
        if (y == WSN - 1) { rL = rs; cL0 = v0; cLL = v63; }
    }
    const int o = (b * WINN + w) * CN + j;
    const int P = BN * WINN * CN;   // 4096
    stats[0*P + o] = T;   stats[1*P + o] = rF;  stats[2*P + o] = rL;
    stats[3*P + o] = cF;  stats[4*P + o] = cL;  stats[5*P + o] = c00;
    stats[6*P + o] = c0L; stats[7*P + o] = cL0; stats[8*P + o] = cLL;
}

// ---------------------------------------------------------------------------
// Kernel 2: batch-independent merged weight pieces.
//   A_w[o,idx] = sum_c fusion_w[o, w*C+c] * conv1_w[(w*C+c), idx]
//   G_w[o,idx] = sum_c fusion_w[o, 4C+c]  * conv1_w[(w*C+c), idx]
//   H_w[o,j*9+hk] = sum_m gk_w[j,w,m] * G_w[o,m*9+hk];   AH = A + H
// ---------------------------------------------------------------------------
__global__ void prep_kernel(const float* __restrict__ conv1_w,
                            const float* __restrict__ gk_w,
                            const float* __restrict__ fusion_w,
                            float* __restrict__ AH) {
    __shared__ float G[KC];
    const int w = blockIdx.x, o = blockIdx.y, t = threadIdx.x;
    const float* Fg = fusion_w + (size_t)o * ((WINN + 1) * CN) + WINN * CN;
    const float* Fw = fusion_w + (size_t)o * ((WINN + 1) * CN) + w * CN;
    for (int idx = t; idx < KC; idx += 256) {
        float acc = 0.f;
        for (int cc = 0; cc < CN; ++cc)
            acc += Fg[cc] * conv1_w[(size_t)(w * CN + cc) * KC + idx];
        G[idx] = acc;
    }
    __syncthreads();
    for (int idx = t; idx < KC; idx += 256) {
        const int j = idx / KHW, hk = idx % KHW;
        float a = 0.f;
        for (int cc = 0; cc < CN; ++cc)
            a += Fw[cc] * conv1_w[(size_t)(w * CN + cc) * KC + idx];
        float h = 0.f;
        const float* gw = gk_w + (size_t)(j * WINN + w) * CN;
        for (int m = 0; m < CN; ++m)
            h += gw[m] * G[m * KHW + hk];
        AH[((size_t)w * CN + o) * KC + idx] = a + h;
    }
}

// ---------------------------------------------------------------------------
// Kernel 3: pooled -> SE MLP -> ww[b,w]  (exact GELU, sigmoid)
// ---------------------------------------------------------------------------
__global__ void ww_kernel(const float* __restrict__ conv1_w,
                          const float* __restrict__ dc_w, const float* __restrict__ dc_b,
                          const float* __restrict__ l1_w, const float* __restrict__ l1_b,
                          const float* __restrict__ l2_w, const float* __restrict__ l2_b,
                          const float* __restrict__ stats, float* __restrict__ ww) {
    __shared__ float S[WINN * CN * KHW];   // shifted sums per (w,j)
    __shared__ float red[WINN][CN];
    __shared__ float h1[4 * WINN];
    const int b = blockIdx.x, t = threadIdx.x;
    const int P = BN * WINN * CN;
    for (int idx = t; idx < WINN * CN; idx += CN) {
        const int w = idx >> 7, j = idx & (CN - 1);
        const int o = (b * WINN + w) * CN + j;
        const float T  = stats[0*P + o], rF = stats[1*P + o], rL = stats[2*P + o];
        const float cF = stats[3*P + o], cL = stats[4*P + o];
        const float c00 = stats[5*P + o], c0L = stats[6*P + o];
        const float cL0 = stats[7*P + o], cLL = stats[8*P + o];
        float* s = S + idx * KHW;
        for (int kh = 0; kh < 3; ++kh)
            for (int kw = 0; kw < 3; ++kw) {
                float v = T;
                if (kh == 0) v -= rL;
                if (kh == 2) v -= rF;
                if (kw == 0) v -= cL;
                if (kw == 2) v -= cF;
                if (kh == 0 && kw == 0) v += cLL;
                if (kh == 0 && kw == 2) v += cL0;
                if (kh == 2 && kw == 0) v += c0L;
                if (kh == 2 && kw == 2) v += c00;
                s[kh * 3 + kw] = v;
            }
    }
    __syncthreads();
    const int c = t;
    for (int w = 0; w < WINN; ++w) {
        const float* wrow = conv1_w + (size_t)(w * CN + c) * KC;
        const float* s = S + w * CN * KHW;
        float acc = 0.f;
        for (int i = 0; i < KC; ++i) acc += wrow[i] * s[i];
        red[w][c] = (acc * (1.0f / (WSN * WSN))) * dc_w[w * CN + c];
    }
    __syncthreads();
    for (int sft = CN / 2; sft >= 1; sft >>= 1) {
        if (c < sft)
            for (int w = 0; w < WINN; ++w) red[w][c] += red[w][c + sft];
        __syncthreads();
    }
    if (t < 4 * WINN) {
        float z = l1_b[t];
        for (int w = 0; w < WINN; ++w)
            z += (red[w][0] + dc_b[w]) * l1_w[t * WINN + w];
        h1[t] = 0.5f * z * (1.0f + erff(z * 0.70710678118654752f));
    }
    __syncthreads();
    if (t < WINN) {
        float z = l2_b[t];
        for (int i = 0; i < 4 * WINN; ++i) z += h1[i] * l2_w[t * 4 * WINN + i];
        ww[b * WINN + t] = 1.0f / (1.0f + __expf(-z));
    }
}

// ---------------------------------------------------------------------------
// Kernel 4: per-sample merged 3x3 kernel, bf16, layout [b][patch][o][j]
//   M_b[o,j,hk] = sum_w ww[b,w]*AH_w[o,j*9+hk] + (sum_c F[o,4C+c]) * gk_b[j]
// ---------------------------------------------------------------------------
__global__ void mkw_kernel(const float* __restrict__ AH, const float* __restrict__ ww,
                           const float* __restrict__ gk_b, const float* __restrict__ fusion_w,
                           bf16_t* __restrict__ Mw) {
    __shared__ float red[CN];
    const int b = blockIdx.x, o = blockIdx.y, t = threadIdx.x;
    red[t] = fusion_w[(size_t)o * ((WINN + 1) * CN) + WINN * CN + t];
    __syncthreads();
    for (int sft = CN / 2; sft >= 1; sft >>= 1) {
        if (t < sft) red[t] += red[t + sft];
        __syncthreads();
    }
    const float c0 = red[0];
    const float w0 = ww[b * WINN + 0], w1 = ww[b * WINN + 1];
    const float w2 = ww[b * WINN + 2], w3 = ww[b * WINN + 3];
    const float* a0 = AH + ((size_t)0 * CN + o) * KC;
    const float* a1 = AH + ((size_t)1 * CN + o) * KC;
    const float* a2 = AH + ((size_t)2 * CN + o) * KC;
    const float* a3 = AH + ((size_t)3 * CN + o) * KC;
    for (int idx = t; idx < KC; idx += CN) {
        const int j = idx / KHW, hk = idx % KHW;
        float v = w0 * a0[idx] + w1 * a1[idx] + w2 * a2[idx] + w3 * a3[idx] + c0 * gk_b[j];
        Mw[(((size_t)b * KHW + hk) * CN + o) * CN + j] = (bf16_t)v;
    }
}

// ---------------------------------------------------------------------------
// Kernel 5: implicit-GEMM 3x3 conv (NHWC), bf16 WMMA, f32 accumulate.
// Weight patches stream via GLOBAL_LOAD_ASYNC_TO_LDS_B128 (ASYNCcnt) into a
// double-buffered LDS region so the copy of patch p+1 overlaps WMMA of patch p.
// Workgroup: 128 threads = 4 waves; tile = 64 x-positions (one row) x 128 o.
// ---------------------------------------------------------------------------
__global__ void __launch_bounds__(128)
conv_wmma_kernel(const float* __restrict__ x, const bf16_t* __restrict__ Mw,
                 const float* __restrict__ fusion_b, float* __restrict__ out) {
    extern __shared__ char smem[];
    bf16_t* xs  = (bf16_t*)smem;                                   // [3][66][128]
    bf16_t* wsh = (bf16_t*)(smem + 3 * 66 * CN * sizeof(bf16_t));  // 2 x [128][128]

    const int x0 = blockIdx.x * 64;
    const int y  = blockIdx.y;
    const int b  = blockIdx.z;
    const int tid = threadIdx.x, wave = tid >> 5, lane = tid & 31;
    const float* xb = x + (size_t)b * LN * CN;
    const bf16_t* MwB = Mw + (size_t)b * KHW * (CN * CN);

    // -- async copy of weight patch p into ping-pong buffer (16 B128 per lane) --
    const unsigned long long mw_base = (unsigned long long)(uintptr_t)MwB;
    const unsigned wsh_lds = (unsigned)(uintptr_t)wsh;   // flat LDS addr[31:0] = offset
    auto issue_patch = [&](int p, int buf) {
        const unsigned long long sbase = mw_base + (unsigned long long)p * (CN * CN * 2);
        const unsigned dbase = wsh_lds + (unsigned)buf * (CN * CN * 2);
        for (int it = 0; it < 16; ++it) {
            const unsigned byte_off = (unsigned)(tid + it * 128) * 16u;
            const unsigned dst = dbase + byte_off;
            asm volatile("global_load_async_to_lds_b128 %0, %1, %2"
                         :
                         : "v"(dst), "v"(byte_off), "s"(sbase)
                         : "memory");
        }
    };

    issue_patch(0, 0);   // prologue: patch 0 in flight while we stage x

    // ---- stage x halo tile (3 x 66 x 128) as bf16; zero-pad out of range ----
    for (int idx = tid; idx < 198 * 32; idx += 128) {
        const int pair = idx >> 5, c4 = (idx & 31) << 2;
        const int r = pair / 66, col = pair % 66;
        const int gy = y + r - 1, gx = x0 + col - 1;
        float4 v = make_float4(0.f, 0.f, 0.f, 0.f);
        if ((unsigned)gy < (unsigned)HN && (unsigned)gx < (unsigned)HN)
            v = *(const float4*)(xb + ((size_t)gy * HN + gx) * CN + c4);
        bf16_t tmp[4] = {(bf16_t)v.x, (bf16_t)v.y, (bf16_t)v.z, (bf16_t)v.w};
        *(uint2*)(xs + (size_t)pair * CN + c4) = *(const uint2*)tmp;
    }

    v8f acc[8] = {};
    const int l16 = lane & 15, lhi = (lane >> 4) & 1;
    const int arow = wave * 16 + l16;            // spatial x offset of this A-row

    for (int p = 0; p < KHW; ++p) {
        // wait for this wave's outstanding async copies, then workgroup barrier:
        // after the barrier every wave's copies into buf[p&1] are complete and
        // every wave has finished computing from buf[(p-1)&1].
        asm volatile("s_wait_asynccnt 0x0" ::: "memory");
        __syncthreads();
        if (p + 1 < KHW) issue_patch(p + 1, (p + 1) & 1);   // overlap with compute

        const bf16_t* wcur = wsh + (size_t)(p & 1) * (CN * CN);
        const int kh = p / 3, kw = p % 3;
        const bf16_t* arow_base = xs + (size_t)(kh * 66 + arow + kw) * CN;

        for (int jc = 0; jc < 4; ++jc) {
            // A fragment (16x32 bf16): lane<16 holds K=j0..j0+7 & j0+16..23,
            // lane>=16 holds K=j0+8..15 & j0+24..31 (per ISA A layout).
            FragA A;
            const bf16_t* ap = arow_base + jc * 32 + lhi * 8;
            A.q[0] = *(const uint4*)(ap);
            A.q[1] = *(const uint4*)(ap + 16);
            for (int n = 0; n < 8; ++n) {
                // B fragment (32x16): lane<16 -> N=lane, K=j0..j0+7 & j0+16..23;
                // lane>=16 -> N=lane-16, K=j0+8..15 & j0+24..31.
                FragB Bf;
                const bf16_t* bp = wcur + (size_t)(n * 16 + l16) * CN + jc * 32 + lhi * 8;
                Bf.q[0] = *(const uint4*)(bp);
                Bf.q[1] = *(const uint4*)(bp + 16);
                acc[n] = __builtin_amdgcn_wmma_f32_16x16x32_bf16(
                    false, A.v, false, Bf.v, (short)0, acc[n], false, false);
            }
        }
        __syncthreads();   // everyone done with buf[p&1] before it is re-filled
    }

    // ---- store: C/D layout -> lane<16: M=vgpr e; lane>=16: M=e+8; N=lane&15 ----
    const int mbase = wave * 16 + (lhi << 3);
    float* outb = out + ((size_t)b * LN + (size_t)y * HN + x0 + mbase) * CN;
    for (int n = 0; n < 8; ++n) {
        const int o = n * 16 + l16;
        const float bias = fusion_b[o];
        for (int e = 0; e < 8; ++e)
            outb[(size_t)e * CN + o] = acc[n][e] + bias;
    }
}

// ---------------------------------------------------------------------------
// Host launcher
// ---------------------------------------------------------------------------
extern "C" void kernel_launch(void* const* d_in, const int* in_sizes, int n_in,
                              void* d_out, int out_size, void* d_ws, size_t ws_size,
                              hipStream_t stream) {
    const float* x        = (const float*)d_in[0];
    const float* conv1_w  = (const float*)d_in[1];
    const float* dc_w     = (const float*)d_in[2];
    const float* dc_b     = (const float*)d_in[3];
    const float* l1_w     = (const float*)d_in[4];
    const float* l1_b     = (const float*)d_in[5];
    const float* l2_w     = (const float*)d_in[6];
    const float* l2_b     = (const float*)d_in[7];
    const float* gk_w     = (const float*)d_in[8];
    const float* gk_b     = (const float*)d_in[9];
    const float* fusion_w = (const float*)d_in[10];
    const float* fusion_b = (const float*)d_in[11];
    float* out = (float*)d_out;

    // workspace carve (floats): stats[9*4096] | ww[32] | AH[4*128*1152] | Mw(bf16)
    float*  wsf   = (float*)d_ws;
    float*  stats = wsf;
    float*  ww    = wsf + 9 * BN * WINN * CN;                 // +36864
    float*  AH    = ww + 32;                                  // +36896
    bf16_t* Mw    = (bf16_t*)(AH + (size_t)WINN * CN * KC);   // +589824 floats

    stats_kernel<<<dim3(WINN, BN), CN, 0, stream>>>(x, stats);
    prep_kernel <<<dim3(WINN, CN), 256, 0, stream>>>(conv1_w, gk_w, fusion_w, AH);
    ww_kernel   <<<BN, CN, 0, stream>>>(conv1_w, dc_w, dc_b, l1_w, l1_b, l2_w, l2_b,
                                        stats, ww);
    mkw_kernel  <<<dim3(BN, CN), CN, 0, stream>>>(AH, ww, gk_b, fusion_w, Mw);

    const size_t shmem = (size_t)(3 * 66 * CN + 2 * CN * CN) * sizeof(bf16_t); // ~116 KB
    conv_wmma_kernel<<<dim3(HN / 64, HN, BN), 128, shmem, stream>>>(x, Mw, fusion_b, out);
}
